// _BasePointnetMSGModule_22067541967442
// MI455X (gfx1250) — matrix operand order repaired and checked by reading, not compiled, not run on hardware
//
#include <hip/hip_runtime.h>
#include <stdint.h>
#include <stddef.h>

// ============================================================================
// PointNet++ MSG set-abstraction for MI455X (gfx1250, wave32, WMMA).
//
// Pipeline:
//   1) fps_kernel          : 1 workgroup/batch, points+dist in registers,
//                            1023 serial argmax steps (latency-optimized).
//   2) per scale:
//      ballquery_kernel    : 1 wave per (b,s); ordered scan -> K smallest
//                            in-ball indices (matches jnp.sort semantics).
//      group_kernel        : gather (pc-center | feat) -> bf16 X0, col-major,
//                            channel-padded to 96.
//      per layer:
//        wprep_kernel      : W fp32 -> bf16, zero-padded to cin_pad.
//        gemm_stats_kernel : WMMA GEMM, per-block partial ch-sums (no Y store).
//        bncoef_kernel     : fixed-order reduction -> BN folded into (a,b).
//        gemm_apply_kernel : WMMA GEMM again, y=relu(a*y+b) -> bf16 X_next,
//        gemm_apply_max_kernel (last layer): + max over K, scatter to d_out.
//
// All matmuls use v_wmma_f32_16x16x32_bf16 (fp32 accumulate). Deterministic:
// no float atomics anywhere (stats reduced in fixed block order).
//
// Workspace layout (bytes), total required ~199 MiB:
//   [0       ) fpsIdx   : 8*1024 i32                =    32768
//   [32768   ) newpts   : 8*1024*3 f32              =    98304
//   [131072  ) coef     : 256 f32 (a | b)           =     1024
//   [133120  ) wslot[3] : 3 * 32768 B bf16 weights  =    98304
//   [231424  ) idxbuf   : 8*1024*64 i32             =  2097152
//   [2328576 ) partial  : 4096 blocks * 256 f32     =  4194304
//   [6522880 ) bufA     : 96ch * 524288 cols bf16   = 100663296
//   [107186176) bufB    : 96ch * 524288 cols bf16   = 100663296
// ============================================================================

typedef __bf16 bf16;
typedef __attribute__((ext_vector_type(8)))  float v8f;
typedef __attribute__((ext_vector_type(8)))  bf16  v8bf;
typedef __attribute__((ext_vector_type(16))) bf16  v16bf;

union AFrag { v16bf v; struct { v8bf lo, hi; } p; };

#define NPTS   16384
#define BATCH  8
#define NPOINT 1024
#define COUT_TOTAL 320

// ---------------- wave32 cross-lane helpers (raw ds_bpermute) ---------------
__device__ __forceinline__ int lane_id() { return (int)(threadIdx.x & 31u); }
__device__ __forceinline__ int bperm(int srclane, int v) {
  return __builtin_amdgcn_ds_bpermute(srclane << 2, v);
}
__device__ __forceinline__ float shfl_xor_f(float v, int mask) {
  return __builtin_bit_cast(float, bperm(lane_id() ^ mask, __builtin_bit_cast(int, v)));
}
__device__ __forceinline__ int shfl_xor_i(int v, int mask) {
  return bperm(lane_id() ^ mask, v);
}
__device__ __forceinline__ int shfl_up_i(int v, int d) {
  int l = lane_id();
  return bperm(l >= d ? l - d : l, v);
}
__device__ __forceinline__ int shfl_bcast_i(int v, int src) { return bperm(src, v); }

// ============================ 1) Farthest point sampling =====================
__global__ __launch_bounds__(1024) void fps_kernel(
    const float* __restrict__ pc, float* __restrict__ pc_sample_out,
    int* __restrict__ fpsIdx, float* __restrict__ newpts)
{
  const int b = blockIdx.x;
  const int t = threadIdx.x;
  const float* px = pc + (size_t)b * 3 * NPTS;
  const float* py = px + NPTS;
  const float* pz = px + 2 * NPTS;

  // 16 points per thread, held entirely in registers (48 + 16 VGPRs).
  float X[16], Y[16], Z[16], D[16];
#pragma unroll
  for (int j = 0; j < 16; ++j) {
    int n = j * 1024 + t;
    X[j] = px[n]; Y[j] = py[n]; Z[j] = pz[n];
    D[j] = 1e10f;
  }

  __shared__ float rv[32];
  __shared__ int   ri[32];
  __shared__ float bc[3];

  float lx = px[0], ly = py[0], lz = pz[0];
  if (t == 0) {
    fpsIdx[b * NPOINT] = 0;
    size_t nb = (size_t)b * NPOINT * 3;
    newpts[nb + 0] = lx; newpts[nb + 1] = ly; newpts[nb + 2] = lz;
    size_t ob = (size_t)b * 3 * NPOINT;
    pc_sample_out[ob] = lx;
    pc_sample_out[ob + NPOINT] = ly;
    pc_sample_out[ob + 2 * NPOINT] = lz;
  }

  const int lane = t & 31, wid = t >> 5;
  for (int i = 1; i < NPOINT; ++i) {
    float best = -1.0f; int bidx = 0;
#pragma unroll
    for (int j = 0; j < 16; ++j) {
      float dx = X[j] - lx, dy = Y[j] - ly, dz = Z[j] - lz;
      float d = fminf(D[j], dx * dx + dy * dy + dz * dz);
      D[j] = d;
      if (d > best) { best = d; bidx = j * 1024 + t; }  // j ascending => first max
    }
    // wave argmax (ties -> smallest index, matching jnp.argmax)
#pragma unroll
    for (int off = 1; off < 32; off <<= 1) {
      float ov = shfl_xor_f(best, off);
      int   oi = shfl_xor_i(bidx, off);
      if (ov > best || (ov == best && oi < bidx)) { best = ov; bidx = oi; }
    }
    if (lane == 0) { rv[wid] = best; ri[wid] = bidx; }
    __syncthreads();
    if (t < 32) {
      best = rv[t]; bidx = ri[t];
#pragma unroll
      for (int off = 1; off < 32; off <<= 1) {
        float ov = shfl_xor_f(best, off);
        int   oi = shfl_xor_i(bidx, off);
        if (ov > best || (ov == best && oi < bidx)) { best = ov; bidx = oi; }
      }
      if (t == 0) {
        float cx = px[bidx], cy = py[bidx], cz = pz[bidx];   // L2-resident
        bc[0] = cx; bc[1] = cy; bc[2] = cz;
        fpsIdx[b * NPOINT + i] = bidx;
        size_t nb = ((size_t)b * NPOINT + i) * 3;
        newpts[nb] = cx; newpts[nb + 1] = cy; newpts[nb + 2] = cz;
        size_t ob = (size_t)b * 3 * NPOINT + i;
        pc_sample_out[ob] = cx;
        pc_sample_out[ob + NPOINT] = cy;
        pc_sample_out[ob + 2 * NPOINT] = cz;
      }
    }
    __syncthreads();
    lx = bc[0]; ly = bc[1]; lz = bc[2];
  }
}

// ============================ 2) Ball query ==================================
// One wave per (b,s). Ordered chunk scan -> K smallest in-ball indices,
// padded with the first hit (center is always in its own ball).
__global__ __launch_bounds__(256) void ballquery_kernel(
    const float* __restrict__ pc, const float* __restrict__ newpts,
    int* __restrict__ idxbuf, float r2, int K)
{
  int gw = (int)((blockIdx.x * blockDim.x + threadIdx.x) >> 5);
  int lane = lane_id();
  if (gw >= BATCH * NPOINT) return;
  int b = gw >> 10;
  const float* px = pc + (size_t)b * 3 * NPTS;
  const float* py = px + NPTS;
  const float* pz = px + 2 * NPTS;
  float cx = newpts[(size_t)gw * 3 + 0];
  float cy = newpts[(size_t)gw * 3 + 1];
  float cz = newpts[(size_t)gw * 3 + 2];
  int* out = idxbuf + (size_t)gw * K;

  int cnt = 0;
  for (int n0 = 0; n0 < NPTS && cnt < K; n0 += 32) {
    int n = n0 + lane;
    float dx = px[n] - cx, dy = py[n] - cy, dz = pz[n] - cz;
    float d = dx * dx + dy * dy + dz * dz;
    int flag = (d <= r2) ? 1 : 0;
    int sc = flag;                        // inclusive prefix sum over lanes
#pragma unroll
    for (int off = 1; off < 32; off <<= 1) {
      int o = shfl_up_i(sc, off);
      if (lane >= off) sc += o;
    }
    int pos = cnt + sc - flag;            // exclusive position
    if (flag && pos < K) out[pos] = n;
    cnt += shfl_bcast_i(sc, 31);
  }
  if (cnt < K) {
    int f = out[0];
    for (int p = cnt + lane; p < K; p += 32) out[p] = f;
  }
}

// ============================ 3) Grouping ====================================
// Build X0 bf16, column-major: col c=(b,s,k) -> 96 contiguous channels
// [pc-center(3) | feat(64) | zero pad(29)].
__global__ __launch_bounds__(256) void group_kernel(
    const float* __restrict__ pc, const float* __restrict__ feat,
    const float* __restrict__ newpts, const int* __restrict__ idxbuf,
    bf16* __restrict__ X0, int K, int ncols)
{
  int c = (int)(blockIdx.x * 256 + threadIdx.x);
  if (c >= ncols) return;
  int bs = c / K;
  int b = bs >> 10;
  int id = idxbuf[c];
  const float* pcb = pc + (size_t)b * 3 * NPTS;
  const float* fb  = feat + (size_t)b * 64 * NPTS;
  bf16* xc = X0 + (size_t)c * 96;
  xc[0] = (bf16)(pcb[id]             - newpts[(size_t)bs * 3 + 0]);
  xc[1] = (bf16)(pcb[id + NPTS]      - newpts[(size_t)bs * 3 + 1]);
  xc[2] = (bf16)(pcb[id + 2 * NPTS]  - newpts[(size_t)bs * 3 + 2]);
#pragma unroll 4
  for (int j = 0; j < 64; ++j) xc[3 + j] = (bf16)fb[(size_t)j * NPTS + id];
#pragma unroll
  for (int j = 67; j < 96; ++j) xc[j] = (bf16)0.0f;
}

// ============================ Weight prep ====================================
__global__ __launch_bounds__(256) void wprep_kernel(
    const float* __restrict__ W, bf16* __restrict__ Wb,
    int cout, int cin, int cinp)
{
  int t = (int)(blockIdx.x * 256 + threadIdx.x);
  if (t >= cout * cinp) return;
  int r = t / cinp, c = t - r * cinp;
  float v = (c < cin) ? W[r * cin + c] : 0.0f;
  Wb[t] = (bf16)v;
}

// ============================ WMMA GEMM core =================================
// D(16x16) = A(16x32 bf16) x B(32x16 bf16) + C(f32), using this code's contiguous k order (same for A and B):
//   A: lane holds row m=lane%16, K = kbase..kbase+15, kbase=(lane<16?0:16)
//      -> one contiguous 32B load from row-major W.
//   B: lane holds col n=lane%16, K = kbase..kbase+15, kbase=(lane<16?0:16)
//      -> one contiguous 32B load from column-major X.
// Each wave owns one 16-row M tile and a 128-column N strip (8 subtiles).
template<int CINP>
__device__ __forceinline__ void gemm_accum(
    const bf16* __restrict__ X, const bf16* __restrict__ Wrow,
    int nbase, int half, int l16, v8f (&acc)[8])
{
#pragma unroll
  for (int kt = 0; kt < CINP / 32; ++kt) {
    AFrag a;
    a.v = *(const v16bf*)(Wrow + kt * 32 + half * 16);
    const int kb = kt * 32 + half * 16;
#pragma unroll
    for (int ns = 0; ns < 8; ++ns) {
      const size_t n = (size_t)(nbase + ns * 16 + l16);
      v16bf bfr = *(const v16bf*)(X + n * CINP + kb);
      acc[ns] = __builtin_amdgcn_wmma_f32_16x16x32_bf16(
          false, a.v, false, bfr, (short)0, acc[ns], false, false);
    }
  }
}

// Pass 1: per-block partial per-channel sum / sumsq (deterministic, no atomics)
template<int CINP, int COUT>
__global__ __launch_bounds__(32 * (COUT / 16)) void gemm_stats_kernel(
    const bf16* __restrict__ X, const bf16* __restrict__ Wb,
    float* __restrict__ partial)
{
  const int lane = lane_id();
  const int wave = (int)(threadIdx.x >> 5);
  const int half = lane >> 4;
  const int l16  = lane & 15;
  const int nbase = (int)blockIdx.x * 128;

  v8f acc[8] = {};
  const bf16* Wrow = Wb + (size_t)(wave * 16 + l16) * CINP;
  gemm_accum<CINP>(X, Wrow, nbase, half, l16, acc);

  float s1[8], s2[8];
#pragma unroll
  for (int v = 0; v < 8; ++v) {
    float a0 = 0.f, a1 = 0.f;
#pragma unroll
    for (int ns = 0; ns < 8; ++ns) { float y = acc[ns][v]; a0 += y; a1 += y * y; }
    s1[v] = a0; s2[v] = a1;
  }
#pragma unroll
  for (int v = 0; v < 8; ++v) {
#pragma unroll
    for (int off = 1; off < 16; off <<= 1) {
      s1[v] += shfl_xor_f(s1[v], off);
      s2[v] += shfl_xor_f(s2[v], off);
    }
  }
  if (l16 == 0) {
    const int ch = wave * 16 + half * 8;
    float* p = partial + (size_t)blockIdx.x * 256;
#pragma unroll
    for (int v = 0; v < 8; ++v) { p[ch + v] = s1[v]; p[128 + ch + v] = s2[v]; }
  }
}

// Fold BN into per-channel affine: coef[c]=a, coef[128+c]=b ; y'=a*y+b
__global__ void bncoef_kernel(
    const float* __restrict__ partial, int nblocks,
    const float* __restrict__ gamma, const float* __restrict__ beta,
    float* __restrict__ coef, int cout, float invn)
{
  int c = (int)threadIdx.x;
  if (c >= cout) return;
  float s = 0.f, q = 0.f;
  for (int blk = 0; blk < nblocks; ++blk) {     // fixed order -> deterministic
    s += partial[(size_t)blk * 256 + c];
    q += partial[(size_t)blk * 256 + 128 + c];
  }
  float mu  = s * invn;
  float var = q * invn - mu * mu;
  float a = gamma[c] * rsqrtf(var + 1e-5f);
  coef[c] = a;
  coef[128 + c] = beta[c] - mu * a;
}

// Pass 2 (mid layer): recompute GEMM, apply affine+ReLU, write bf16 X_next
template<int CINP, int COUT>
__global__ __launch_bounds__(32 * (COUT / 16)) void gemm_apply_kernel(
    const bf16* __restrict__ X, const bf16* __restrict__ Wb,
    const float* __restrict__ coef, bf16* __restrict__ Xn)
{
  const int lane = lane_id();
  const int wave = (int)(threadIdx.x >> 5);
  const int half = lane >> 4;
  const int l16  = lane & 15;
  const int nbase = (int)blockIdx.x * 128;

  v8f acc[8] = {};
  const bf16* Wrow = Wb + (size_t)(wave * 16 + l16) * CINP;
  gemm_accum<CINP>(X, Wrow, nbase, half, l16, acc);

  const int ch0 = wave * 16 + half * 8;
  float ca[8], cb[8];
#pragma unroll
  for (int v = 0; v < 8; ++v) { ca[v] = coef[ch0 + v]; cb[v] = coef[128 + ch0 + v]; }
#pragma unroll
  for (int ns = 0; ns < 8; ++ns) {
    v8bf o;
#pragma unroll
    for (int v = 0; v < 8; ++v) {
      float y = fmaxf(acc[ns][v] * ca[v] + cb[v], 0.0f);
      o[v] = (bf16)y;
    }
    const size_t n = (size_t)(nbase + ns * 16 + l16);
    *(v8bf*)(Xn + n * COUT + ch0) = o;          // 16B aligned col-major store
  }
}

// Pass 2 (last layer): affine+ReLU fused with max-over-K + scatter to output
template<int CINP, int COUT, int K>
__global__ __launch_bounds__(32 * (COUT / 16)) void gemm_apply_max_kernel(
    const bf16* __restrict__ X, const bf16* __restrict__ Wb,
    const float* __restrict__ coef, float* __restrict__ out2, int cbase)
{
  const int lane = lane_id();
  const int wave = (int)(threadIdx.x >> 5);
  const int half = lane >> 4;
  const int l16  = lane & 15;
  const int nbase = (int)blockIdx.x * 128;

  v8f acc[8] = {};
  const bf16* Wrow = Wb + (size_t)(wave * 16 + l16) * CINP;
  gemm_accum<CINP>(X, Wrow, nbase, half, l16, acc);

  const int ch0 = wave * 16 + half * 8;
  float ca[8], cb[8];
#pragma unroll
  for (int v = 0; v < 8; ++v) { ca[v] = coef[ch0 + v]; cb[v] = coef[128 + ch0 + v]; }

  constexpr int SPG = K / 16;   // N-subtiles per (b,s) group
  constexpr int GPT = 8 / SPG;  // groups per block strip
#pragma unroll
  for (int g = 0; g < GPT; ++g) {
    float mv[8];
#pragma unroll
    for (int v = 0; v < 8; ++v) mv[v] = 0.0f;   // relu floor
#pragma unroll
    for (int st = 0; st < SPG; ++st) {
      const int ns = g * SPG + st;
#pragma unroll
      for (int v = 0; v < 8; ++v)
        mv[v] = fmaxf(mv[v], acc[ns][v] * ca[v] + cb[v]);
    }
#pragma unroll
    for (int off = 1; off < 16; off <<= 1) {
#pragma unroll
      for (int v = 0; v < 8; ++v)
        mv[v] = fmaxf(mv[v], shfl_xor_f(mv[v], off));
    }
    if (l16 == 0) {
      const int col0 = nbase + g * K;
      const int bs = col0 / K;
      const int b = bs >> 10;
      const int s = bs & 1023;
      float* o = out2 + ((size_t)b * COUT_TOTAL + (cbase + ch0)) * NPOINT + s;
#pragma unroll
      for (int v = 0; v < 8; ++v) o[(size_t)v * NPOINT] = mv[v];
    }
  }
}

// ============================ Host-side orchestration ========================
static inline void prep_w(const float* W, bf16* Wb, int cout, int cin, int cinp,
                          hipStream_t s) {
  int n = cout * cinp;
  wprep_kernel<<<(n + 255) / 256, 256, 0, s>>>(W, Wb, cout, cin, cinp);
}

template<int CINP, int COUT>
static inline void run_mid(const bf16* X, const bf16* Wb, float* partial,
                           float* coef, const float* gamma, const float* beta,
                           bf16* Xn, int ncols, hipStream_t stream) {
  int nblk = ncols / 128;
  dim3 blk(32 * (COUT / 16));
  gemm_stats_kernel<CINP, COUT><<<nblk, blk, 0, stream>>>(X, Wb, partial);
  bncoef_kernel<<<1, 128, 0, stream>>>(partial, nblk, gamma, beta, coef, COUT,
                                       1.0f / (float)ncols);
  gemm_apply_kernel<CINP, COUT><<<nblk, blk, 0, stream>>>(X, Wb, coef, Xn);
}

template<int CINP, int COUT, int K>
static inline void run_final(const bf16* X, const bf16* Wb, float* partial,
                             float* coef, const float* gamma, const float* beta,
                             float* out2, int cbase, int ncols,
                             hipStream_t stream) {
  int nblk = ncols / 128;
  dim3 blk(32 * (COUT / 16));
  gemm_stats_kernel<CINP, COUT><<<nblk, blk, 0, stream>>>(X, Wb, partial);
  bncoef_kernel<<<1, 128, 0, stream>>>(partial, nblk, gamma, beta, coef, COUT,
                                       1.0f / (float)ncols);
  gemm_apply_max_kernel<CINP, COUT, K><<<nblk, blk, 0, stream>>>(X, Wb, coef,
                                                                 out2, cbase);
}

extern "C" void kernel_launch(void* const* d_in, const int* in_sizes, int n_in,
                              void* d_out, int out_size, void* d_ws,
                              size_t ws_size, hipStream_t stream) {
  (void)in_sizes; (void)n_in; (void)out_size; (void)ws_size;

  const float* pc   = (const float*)d_in[0];
  const float* feat = (const float*)d_in[1];
  float* out  = (float*)d_out;
  float* out2 = out + (size_t)BATCH * 3 * NPOINT;   // after pc_sample

  char* ws = (char*)d_ws;
  int*   fpsIdx   = (int*)(ws + 0);
  float* newpts   = (float*)(ws + 32768);
  float* coef     = (float*)(ws + 131072);
  bf16*  wslot0   = (bf16*)(ws + 133120);
  bf16*  wslot1   = (bf16*)(ws + 133120 + 32768);
  bf16*  wslot2   = (bf16*)(ws + 133120 + 65536);
  int*   idxbuf   = (int*)(ws + 231424);
  float* partial  = (float*)(ws + 2328576);
  bf16*  bufA     = (bf16*)(ws + 6522880);
  bf16*  bufB     = (bf16*)(ws + 107186176);
  (void)fpsIdx;

  // ---- FPS + sampled coords (writes pc_sample part of d_out) ----
  fps_kernel<<<BATCH, 1024, 0, stream>>>(pc, out, fpsIdx, newpts);

  // param layout in d_in: base = 2 + 9*scale + 3*layer -> {W, gamma, beta}
  #define PW(s, l) ((const float*)d_in[2 + 9 * (s) + 3 * (l) + 0])
  #define PG(s, l) ((const float*)d_in[2 + 9 * (s) + 3 * (l) + 1])
  #define PB(s, l) ((const float*)d_in[2 + 9 * (s) + 3 * (l) + 2])

  // ---------------- Scale 0: r=0.1, K=16, 67->32->32->64 ----------------
  {
    const int K = 16, ncols = BATCH * NPOINT * K;                 // 131072
    ballquery_kernel<<<1024, 256, 0, stream>>>(pc, newpts, idxbuf, 0.01f, K);
    group_kernel<<<ncols / 256, 256, 0, stream>>>(pc, feat, newpts, idxbuf,
                                                  bufA, K, ncols);
    prep_w(PW(0, 0), wslot0, 32, 67, 96, stream);
    prep_w(PW(0, 1), wslot1, 32, 32, 32, stream);
    prep_w(PW(0, 2), wslot2, 64, 32, 32, stream);
    run_mid<96, 32>(bufA, wslot0, partial, coef, PG(0, 0), PB(0, 0), bufB,
                    ncols, stream);
    run_mid<32, 32>(bufB, wslot1, partial, coef, PG(0, 1), PB(0, 1), bufA,
                    ncols, stream);
    run_final<32, 64, 16>(bufA, wslot2, partial, coef, PG(0, 2), PB(0, 2),
                          out2, 0, ncols, stream);
  }

  // ---------------- Scale 1: r=0.2, K=32, 67->64->64->128 ----------------
  {
    const int K = 32, ncols = BATCH * NPOINT * K;                 // 262144
    ballquery_kernel<<<1024, 256, 0, stream>>>(pc, newpts, idxbuf, 0.04f, K);
    group_kernel<<<ncols / 256, 256, 0, stream>>>(pc, feat, newpts, idxbuf,
                                                  bufA, K, ncols);
    prep_w(PW(1, 0), wslot0, 64, 67, 96, stream);
    prep_w(PW(1, 1), wslot1, 64, 64, 64, stream);
    prep_w(PW(1, 2), wslot2, 128, 64, 64, stream);
    run_mid<96, 64>(bufA, wslot0, partial, coef, PG(1, 0), PB(1, 0), bufB,
                    ncols, stream);
    run_mid<64, 64>(bufB, wslot1, partial, coef, PG(1, 1), PB(1, 1), bufA,
                    ncols, stream);
    run_final<64, 128, 32>(bufA, wslot2, partial, coef, PG(1, 2), PB(1, 2),
                           out2, 64, ncols, stream);
  }

  // ---------------- Scale 2: r=0.4, K=64, 67->96->96->128 ----------------
  {
    const int K = 64, ncols = BATCH * NPOINT * K;                 // 524288
    ballquery_kernel<<<1024, 256, 0, stream>>>(pc, newpts, idxbuf, 0.16f, K);
    group_kernel<<<ncols / 256, 256, 0, stream>>>(pc, feat, newpts, idxbuf,
                                                  bufA, K, ncols);
    prep_w(PW(2, 0), wslot0, 96, 67, 96, stream);
    prep_w(PW(2, 1), wslot1, 96, 96, 96, stream);
    prep_w(PW(2, 2), wslot2, 128, 96, 96, stream);
    run_mid<96, 96>(bufA, wslot0, partial, coef, PG(2, 0), PB(2, 0), bufB,
                    ncols, stream);
    run_mid<96, 96>(bufB, wslot1, partial, coef, PG(2, 1), PB(2, 1), bufA,
                    ncols, stream);
    run_final<96, 128, 64>(bufA, wslot2, partial, coef, PG(2, 2), PB(2, 2),
                           out2, 192, ncols, stream);
  }

  #undef PW
  #undef PG
  #undef PB
}
